// LucidRainsMinimal_buggy_360777253456
// MI455X (gfx1250) — matrix operand order repaired and checked by reading, not compiled, not run on hardware
//
#include <hip/hip_runtime.h>
#include <hip/hip_bf16.h>

typedef __attribute__((ext_vector_type(16))) _Float16 v16h;
typedef __attribute__((ext_vector_type(8)))  _Float16 v8h;
typedef __attribute__((ext_vector_type(4)))  _Float16 v4h;
typedef __attribute__((ext_vector_type(8)))  float    v8f;

#define NPTS 8192
#define DIM  512
#define HKVN 2
#define GN   4
#define DH   64
#define MSZ  64
#define NBLK 128
#define HID  4096
#define ATT_SCALE 0.125f   // 64^-0.5

#define BM 128
#define BN 64
#define BK 32

#define WMMA_F16(A_, B_, C_) \
    __builtin_amdgcn_wmma_f32_16x16x32_f16(false, (A_), false, (B_), (short)0, (C_), false, false)

// ---------------------------------------------------------------------------
// WMMA fragment helpers (contiguous k = 16h + s order in A and B, 16-bit data).
// LDS layouts are chosen so each fragment is exactly 2x ds_load_b128:
//   A tile : row-major [m][k], row stride multiple of 8 halves (16B)
//   B tile : TRANSPOSED [n][k] ("k-contiguous"), same stride rule
// ---------------------------------------------------------------------------
__device__ __forceinline__ v16h load_a_frag(const _Float16* base, int stride,
                                            int row0, int k0, int lane) {
    const int m  = row0 + (lane & 15);
    const int hi = lane >> 4;
    const _Float16* p = base + m * stride + k0 + hi * 16;
    union { v16h v; v8h h[2]; } u;
    u.h[0] = *(const v8h*)(p);        // K = k0+hi*16 .. +7
    u.h[1] = *(const v8h*)(p + 8);    // K = k0+hi*16+8 .. +15
    return u.v;
}

__device__ __forceinline__ v16h load_b_frag_t(const _Float16* baseT, int stride,
                                              int k0, int n0, int lane) {
    const int n  = n0 + (lane & 15);
    const int kb = k0 + (lane >> 4) * 16;
    const _Float16* p = baseT + n * stride + kb;
    union { v16h v; v8h h[2]; } u;
    u.h[0] = *(const v8h*)(p);
    u.h[1] = *(const v8h*)(p + 8);
    return u.v;
}

__device__ __forceinline__ v4h cvt4(float4 f) {
    v4h r = { (_Float16)f.x, (_Float16)f.y, (_Float16)f.z, (_Float16)f.w };
    return r;
}

// ---------------------------------------------------------------------------
// Generic GEMM: C[M,N] = act( scale * A[M,K] @ op(B) + bias ), f16 WMMA path.
// Block = 128 threads (4 waves); tile 128x64; each wave owns a 32x64 strip.
// All 12 fragment loads are issued before the 8 WMMAs of a K-step so the
// LDS waits coalesce and the matrix pipe runs back-to-back.
// ---------------------------------------------------------------------------
__global__ __launch_bounds__(128) void gemm_wmma_f16(
    const float* __restrict__ A, int lda,
    const float* __restrict__ B, int ldb, int transB,
    const float* __restrict__ bias,
    float* __restrict__ C, int ldc,
    int Md, int Nd, int Kd, int relu, float scale)
{
    __shared__ alignas(16) _Float16 As[BM][40];   // [m][k], 80B rows
    __shared__ alignas(16) _Float16 Bt[BN][40];   // [n][k] transposed tile
    const int tid  = threadIdx.x;
    const int lane = tid & 31;
    const int wv   = tid >> 5;
    const int tM   = blockIdx.y * BM;
    const int tN   = blockIdx.x * BN;
    const bool interior = (tM + BM <= Md) && (tN + BN <= Nd) &&
                          ((Kd & (BK - 1)) == 0) && ((lda & 3) == 0) && ((ldb & 3) == 0);

    v8f acc[2][4] = {};

    for (int k0 = 0; k0 < Kd; k0 += BK) {
        if (interior) {
            for (int i = tid; i < BM * (BK / 4); i += 128) {
                int r = i >> 3, c4 = (i & 7) * 4;
                float4 f = *(const float4*)&A[(long)(tM + r) * lda + k0 + c4];
                *(v4h*)&As[r][c4] = cvt4(f);
            }
            if (transB) {
                for (int i = tid; i < BN * (BK / 4); i += 128) {
                    int n = i >> 3, c4 = (i & 7) * 4;
                    float4 f = *(const float4*)&B[(long)(tN + n) * ldb + k0 + c4];
                    *(v4h*)&Bt[n][c4] = cvt4(f);
                }
            } else {
                for (int i = tid; i < BK * (BN / 4); i += 128) {
                    int r = i >> 4, n4 = (i & 15) * 4;
                    float4 f = *(const float4*)&B[(long)(k0 + r) * ldb + tN + n4];
                    Bt[n4 + 0][r] = (_Float16)f.x;
                    Bt[n4 + 1][r] = (_Float16)f.y;
                    Bt[n4 + 2][r] = (_Float16)f.z;
                    Bt[n4 + 3][r] = (_Float16)f.w;
                }
            }
        } else {
            for (int i = tid; i < BM * BK; i += 128) {
                int r = i >> 5, c = i & 31;
                int gr = tM + r, gc = k0 + c;
                As[r][c] = (_Float16)((gr < Md && gc < Kd) ? A[(long)gr * lda + gc] : 0.0f);
            }
            if (transB) {
                for (int i = tid; i < BN * BK; i += 128) {
                    int c = i >> 5, r = i & 31;
                    int gk = k0 + r, gn = tN + c;
                    Bt[c][r] = (_Float16)((gk < Kd && gn < Nd) ? B[(long)gn * ldb + gk] : 0.0f);
                }
            } else {
                for (int i = tid; i < BK * BN; i += 128) {
                    int r = i >> 6, c = i & 63;
                    int gk = k0 + r, gn = tN + c;
                    Bt[c][r] = (_Float16)((gk < Kd && gn < Nd) ? B[(long)gk * ldb + gn] : 0.0f);
                }
            }
        }
        if (k0 + BK < Kd) {
            int pr = tM + tid;
            if (pr < Md) __builtin_prefetch(&A[(long)pr * lda + k0 + BK], 0, 1);
            if (!transB) {
                int pk = k0 + BK + (tid & 31);
                int pn = tN + ((tid >> 5) << 4);
                if (pk < Kd && pn < Nd) __builtin_prefetch(&B[(long)pk * ldb + pn], 0, 1);
            }
        }
        __syncthreads();

        // issue all fragment loads first, then all 8 WMMAs
        v16h af0 = load_a_frag(&As[0][0], 40, wv * 32, 0, lane);
        v16h af1 = load_a_frag(&As[0][0], 40, wv * 32 + 16, 0, lane);
        v16h bf0 = load_b_frag_t(&Bt[0][0], 40, 0, 0, lane);
        v16h bf1 = load_b_frag_t(&Bt[0][0], 40, 0, 16, lane);
        v16h bf2 = load_b_frag_t(&Bt[0][0], 40, 0, 32, lane);
        v16h bf3 = load_b_frag_t(&Bt[0][0], 40, 0, 48, lane);
        acc[0][0] = WMMA_F16(af0, bf0, acc[0][0]);
        acc[1][0] = WMMA_F16(af1, bf0, acc[1][0]);
        acc[0][1] = WMMA_F16(af0, bf1, acc[0][1]);
        acc[1][1] = WMMA_F16(af1, bf1, acc[1][1]);
        acc[0][2] = WMMA_F16(af0, bf2, acc[0][2]);
        acc[1][2] = WMMA_F16(af1, bf2, acc[1][2]);
        acc[0][3] = WMMA_F16(af0, bf3, acc[0][3]);
        acc[1][3] = WMMA_F16(af1, bf3, acc[1][3]);
        __syncthreads();
    }

#pragma unroll
    for (int s = 0; s < 2; ++s) {
#pragma unroll
        for (int c = 0; c < 4; ++c) {
            int gn = tN + c * 16 + (lane & 15);
            if (gn >= Nd) continue;
            float bv = bias ? bias[gn] : 0.0f;
#pragma unroll
            for (int r = 0; r < 8; ++r) {
                int gm = tM + wv * 32 + s * 16 + r + (lane >> 4) * 8;
                if (gm >= Md) continue;
                float v = acc[s][c][r] * scale + bv;
                if (relu) v = fmaxf(v, 0.0f);
                C[(long)gm * ldc + gn] = v;
            }
        }
    }
}

// ---------------------------------------------------------------------------
// Ball-relative position embedding: h = x + rel @ pe_w + pe_b
// ---------------------------------------------------------------------------
__global__ __launch_bounds__(256) void posemb_kernel(
    const float* __restrict__ x, const float* __restrict__ pos,
    const float* __restrict__ pe_w, const float* __restrict__ pe_b,
    float* __restrict__ h)
{
    const int ball = blockIdx.x;
    const int tid  = threadIdx.x;
    __shared__ float rp[64][3];
    __shared__ float cm[3];
    if (tid < 64) {
#pragma unroll
        for (int j = 0; j < 3; ++j) rp[tid][j] = pos[(ball * 64 + tid) * 3 + j];
    }
    __syncthreads();
    if (tid < 3) {
        float s = 0.0f;
        for (int m = 0; m < 64; ++m) s += rp[m][tid];
        cm[tid] = s * (1.0f / 64.0f);
    }
    __syncthreads();
    for (int i = tid; i < 64 * DIM; i += 256) {
        int m = i >> 9, d = i & 511;
        long n = (long)ball * 64 + m;
        float r0 = rp[m][0] - cm[0], r1 = rp[m][1] - cm[1], r2 = rp[m][2] - cm[2];
        h[n * DIM + d] = x[n * DIM + d] + r0 * pe_w[d] + r1 * pe_w[512 + d] +
                         r2 * pe_w[1024 + d] + pe_b[d];
    }
}

// ---------------------------------------------------------------------------
// Build compress-MLP input: (kv block + per-block pos emb) -> [HKV, NB, HID]
// ---------------------------------------------------------------------------
__global__ __launch_bounds__(256) void build_cin(
    const float* __restrict__ kv, const float* __restrict__ pe,
    float* __restrict__ out)
{
    long i = (long)blockIdx.x * 256 + threadIdx.x;
    if (i >= (long)HKVN * NBLK * HID) return;
    int d  = (int)(i & 63);
    int m  = (int)((i >> 6) & 63);
    int b  = (int)((i >> 12) & 127);
    int hv = (int)(i >> 19);
    out[i] = kv[((long)(b * 64 + m)) * 128 + hv * 64 + d] + pe[(hv * 64 + m) * 64 + d];
}

// ---------------------------------------------------------------------------
// Softmax over NB for all G, importance mean, top-2 selection (in-place).
// ---------------------------------------------------------------------------
__global__ __launch_bounds__(128) void softmax_sel_kernel(
    float* __restrict__ csim, int* __restrict__ sel)
{
    const int hkv = blockIdx.x >> 13;
    const int n   = blockIdx.x & 8191;
    const int tid = threadIdx.x;
    __shared__ float buf[128];
    __shared__ float impRow[128];
    impRow[tid] = 0.0f;
    for (int g = 0; g < GN; ++g) {
        float* row = csim + (((long)(hkv * GN + g) * NPTS) + n) * NBLK;
        float v = row[tid];
        buf[tid] = v; __syncthreads();
        for (int s = 64; s > 0; s >>= 1) {
            if (tid < s) buf[tid] = fmaxf(buf[tid], buf[tid + s]);
            __syncthreads();
        }
        float mx = buf[0]; __syncthreads();
        float e = __expf(v - mx);
        buf[tid] = e; __syncthreads();
        for (int s = 64; s > 0; s >>= 1) {
            if (tid < s) buf[tid] += buf[tid + s];
            __syncthreads();
        }
        float a = e / buf[0]; __syncthreads();
        row[tid] = a;
        impRow[tid] += a * 0.25f;
    }
    __syncthreads();
    if (tid == 0) {
        int i0 = 0; float v0 = -1e30f;
        for (int j = 0; j < NBLK; ++j) if (impRow[j] > v0) { v0 = impRow[j]; i0 = j; }
        int i1 = 0; float v1 = -1e30f;
        for (int j = 0; j < NBLK; ++j) if (j != i0 && impRow[j] > v1) { v1 = impRow[j]; i1 = j; }
        sel[((long)hkv * NPTS + n) * 2 + 0] = i0;
        sel[((long)hkv * NPTS + n) * 2 + 1] = i1;
    }
}

// ---------------------------------------------------------------------------
// Fine branch: per-query top-2 block gather attention (irregular -> VALU).
// ---------------------------------------------------------------------------
__global__ __launch_bounds__(128) void fine_kernel(
    const float* __restrict__ q, const float* __restrict__ kbuf,
    const float* __restrict__ vbuf, const int* __restrict__ sel,
    float* __restrict__ fine)
{
    const int hkv = blockIdx.x >> 13;
    const int n   = blockIdx.x & 8191;
    const int tid = threadIdx.x;
    __shared__ float qs[GN][DH];
    __shared__ float sc[GN][128];
    __shared__ int   blks[2];
    if (tid < 2) blks[tid] = sel[((long)hkv * NPTS + n) * 2 + tid];
    for (int i = tid; i < GN * DH; i += 128) {
        int g = i >> 6, d = i & 63;
        qs[g][d] = q[(long)n * 512 + hkv * 256 + g * 64 + d];
    }
    __syncthreads();
    const int t = tid >> 6, m = tid & 63;
    const long krow = (long)(blks[t] * 64 + m) * 128 + hkv * 64;
    float s0 = 0, s1 = 0, s2 = 0, s3 = 0;
    for (int d = 0; d < DH; ++d) {
        float kv = kbuf[krow + d];
        s0 += qs[0][d] * kv; s1 += qs[1][d] * kv;
        s2 += qs[2][d] * kv; s3 += qs[3][d] * kv;
    }
    sc[0][tid] = s0 * ATT_SCALE; sc[1][tid] = s1 * ATT_SCALE;
    sc[2][tid] = s2 * ATT_SCALE; sc[3][tid] = s3 * ATT_SCALE;
    __syncthreads();
    if (tid < GN) {
        float mx = -1e30f;
        for (int j = 0; j < 128; ++j) mx = fmaxf(mx, sc[tid][j]);
        float s = 0.0f;
        for (int j = 0; j < 128; ++j) { float e = __expf(sc[tid][j] - mx); sc[tid][j] = e; s += e; }
        float inv = 1.0f / s;
        for (int j = 0; j < 128; ++j) sc[tid][j] *= inv;
    }
    __syncthreads();
    for (int i = tid; i < GN * DH; i += 128) {
        int g = i >> 6, d = i & 63;
        float o = 0.0f;
        for (int j = 0; j < 128; ++j) {
            long vrow = (long)(blks[j >> 6] * 64 + (j & 63)) * 128 + hkv * 64;
            o += sc[g][j] * vbuf[vrow + d];
        }
        fine[(((long)(hkv * GN + g) * NPTS) + n) * DH + d] = o;
    }
}

// ---------------------------------------------------------------------------
// Local branch: per-ball 64x64 attention, full WMMA (S=QK^T, softmax, O=PV).
// ---------------------------------------------------------------------------
__global__ __launch_bounds__(128) void local_attn_wmma(
    const float* __restrict__ q, const float* __restrict__ kbuf,
    const float* __restrict__ vbuf, float* __restrict__ loc)
{
    const int hkv  = blockIdx.x >> 7;
    const int ball = blockIdx.x & 127;
    const int tid  = threadIdx.x;
    const int lane = tid & 31;
    const int wv   = tid >> 5;
    __shared__ alignas(16) _Float16 Qs[64][72];   // Q ([m][k]), reused as P ([m][j])
    __shared__ alignas(16) _Float16 Ks[64][72];   // K ([j][d]) == B^T for S
    __shared__ alignas(16) _Float16 Vt[64][72];   // V^T ([d][j]) == B^T for O
    __shared__ float Ss[64][65];

    for (int i = tid; i < 64 * 16; i += 128) {
        int j = i >> 4, d4 = (i & 15) * 4;
        long row = (long)(ball * 64 + j) * 128 + hkv * 64 + d4;
        float4 kf = *(const float4*)&kbuf[row];
        float4 vf = *(const float4*)&vbuf[row];
        *(v4h*)&Ks[j][d4] = cvt4(kf);
        Vt[d4 + 0][j] = (_Float16)vf.x;
        Vt[d4 + 1][j] = (_Float16)vf.y;
        Vt[d4 + 2][j] = (_Float16)vf.z;
        Vt[d4 + 3][j] = (_Float16)vf.w;
    }

    for (int g = 0; g < GN; ++g) {
        __syncthreads();
        for (int i = tid; i < 64 * 16; i += 128) {
            int m = i >> 4, d4 = (i & 15) * 4;
            float4 f = *(const float4*)&q[(long)(ball * 64 + m) * 512 + hkv * 256 + g * 64 + d4];
            *(v4h*)&Qs[m][d4] = cvt4(f);
        }
        __syncthreads();

        v8f acc[4] = {};
        {
            v16h a0 = load_a_frag(&Qs[0][0], 72, wv * 16, 0, lane);
            v16h a1 = load_a_frag(&Qs[0][0], 72, wv * 16, 32, lane);
            v16h b00 = load_b_frag_t(&Ks[0][0], 72, 0, 0, lane);
            v16h b01 = load_b_frag_t(&Ks[0][0], 72, 0, 16, lane);
            v16h b02 = load_b_frag_t(&Ks[0][0], 72, 0, 32, lane);
            v16h b03 = load_b_frag_t(&Ks[0][0], 72, 0, 48, lane);
            v16h b10 = load_b_frag_t(&Ks[0][0], 72, 32, 0, lane);
            v16h b11 = load_b_frag_t(&Ks[0][0], 72, 32, 16, lane);
            v16h b12 = load_b_frag_t(&Ks[0][0], 72, 32, 32, lane);
            v16h b13 = load_b_frag_t(&Ks[0][0], 72, 32, 48, lane);
            acc[0] = WMMA_F16(a0, b00, acc[0]); acc[0] = WMMA_F16(a1, b10, acc[0]);
            acc[1] = WMMA_F16(a0, b01, acc[1]); acc[1] = WMMA_F16(a1, b11, acc[1]);
            acc[2] = WMMA_F16(a0, b02, acc[2]); acc[2] = WMMA_F16(a1, b12, acc[2]);
            acc[3] = WMMA_F16(a0, b03, acc[3]); acc[3] = WMMA_F16(a1, b13, acc[3]);
        }
#pragma unroll
        for (int c = 0; c < 4; ++c) {
            int n = c * 16 + (lane & 15);
#pragma unroll
            for (int r = 0; r < 8; ++r) {
                int m = wv * 16 + r + (lane >> 4) * 8;
                Ss[m][n] = acc[c][r] * ATT_SCALE;
            }
        }
        __syncthreads();

        if (tid < 64) {
            float mx = -1e30f;
            for (int j = 0; j < 64; ++j) mx = fmaxf(mx, Ss[tid][j]);
            float s = 0.0f;
            for (int j = 0; j < 64; ++j) { float e = __expf(Ss[tid][j] - mx); Ss[tid][j] = e; s += e; }
            float inv = 1.0f / s;
            for (int j = 0; j < 64; ++j) Qs[tid][j] = (_Float16)(Ss[tid][j] * inv);
        }
        __syncthreads();

        v8f oacc[4] = {};
        {
            v16h a0 = load_a_frag(&Qs[0][0], 72, wv * 16, 0, lane);
            v16h a1 = load_a_frag(&Qs[0][0], 72, wv * 16, 32, lane);
            v16h b00 = load_b_frag_t(&Vt[0][0], 72, 0, 0, lane);
            v16h b01 = load_b_frag_t(&Vt[0][0], 72, 0, 16, lane);
            v16h b02 = load_b_frag_t(&Vt[0][0], 72, 0, 32, lane);
            v16h b03 = load_b_frag_t(&Vt[0][0], 72, 0, 48, lane);
            v16h b10 = load_b_frag_t(&Vt[0][0], 72, 32, 0, lane);
            v16h b11 = load_b_frag_t(&Vt[0][0], 72, 32, 16, lane);
            v16h b12 = load_b_frag_t(&Vt[0][0], 72, 32, 32, lane);
            v16h b13 = load_b_frag_t(&Vt[0][0], 72, 32, 48, lane);
            oacc[0] = WMMA_F16(a0, b00, oacc[0]); oacc[0] = WMMA_F16(a1, b10, oacc[0]);
            oacc[1] = WMMA_F16(a0, b01, oacc[1]); oacc[1] = WMMA_F16(a1, b11, oacc[1]);
            oacc[2] = WMMA_F16(a0, b02, oacc[2]); oacc[2] = WMMA_F16(a1, b12, oacc[2]);
            oacc[3] = WMMA_F16(a0, b03, oacc[3]); oacc[3] = WMMA_F16(a1, b13, oacc[3]);
        }
#pragma unroll
        for (int c = 0; c < 4; ++c) {
            int d = c * 16 + (lane & 15);
#pragma unroll
            for (int r = 0; r < 8; ++r) {
                int m = wv * 16 + r + (lane >> 4) * 8;
                loc[(((long)(hkv * GN + g) * NPTS) + ball * 64 + m) * DH + d] = oacc[c][r];
            }
        }
    }
}

// ---------------------------------------------------------------------------
// Gated combine of the three branches into [N, HQ*DH]
// ---------------------------------------------------------------------------
__global__ __launch_bounds__(256) void combine_kernel(
    const float* __restrict__ gpre, const float* __restrict__ comp,
    const float* __restrict__ fine, const float* __restrict__ loc,
    float* __restrict__ combined)
{
    long i = (long)blockIdx.x * 256 + threadIdx.x;
    if (i >= (long)NPTS * 512) return;
    int d   = (int)(i & 63);
    int g   = (int)((i >> 6) & 3);
    int hkv = (int)((i >> 8) & 1);
    long n  = i >> 9;
    float g0 = 1.0f / (1.0f + __expf(-gpre[n * 24 + 0 * 8 + hkv * 4 + g]));
    float g1 = 1.0f / (1.0f + __expf(-gpre[n * 24 + 1 * 8 + hkv * 4 + g]));
    float g2 = 1.0f / (1.0f + __expf(-gpre[n * 24 + 2 * 8 + hkv * 4 + g]));
    long hg = ((long)(hkv * GN + g) * NPTS + n) * DH + d;
    combined[i] = g0 * comp[hg] + g1 * fine[hg] + g2 * loc[hg];
}

// ---------------------------------------------------------------------------
static inline void launch_gemm(const float* A, int lda, const float* B, int ldb,
                               int transB, const float* bias, float* C, int ldc,
                               int M, int N, int K, int relu, float scale,
                               hipStream_t s)
{
    dim3 grid((N + BN - 1) / BN, (M + BM - 1) / BM), blk(128);
    gemm_wmma_f16<<<grid, blk, 0, s>>>(A, lda, B, ldb, transB, bias, C, ldc,
                                       M, N, K, relu, scale);
}

extern "C" void kernel_launch(void* const* d_in, const int* in_sizes, int n_in,
                              void* d_out, int out_size, void* d_ws, size_t ws_size,
                              hipStream_t stream)
{
    const float* x     = (const float*)d_in[0];
    const float* pos   = (const float*)d_in[1];
    const float* pe_w  = (const float*)d_in[2];
    const float* pe_b  = (const float*)d_in[3];
    const float* wq    = (const float*)d_in[4];
    const float* wk    = (const float*)d_in[5];
    const float* wvw   = (const float*)d_in[6];
    const float* k_pos = (const float*)d_in[7];
    const float* v_pos = (const float*)d_in[8];
    const float* ck_w1 = (const float*)d_in[9];
    const float* ck_b1 = (const float*)d_in[10];
    const float* ck_w2 = (const float*)d_in[11];
    const float* ck_b2 = (const float*)d_in[12];
    const float* cv_w1 = (const float*)d_in[13];
    const float* cv_b1 = (const float*)d_in[14];
    const float* cv_w2 = (const float*)d_in[15];
    const float* cv_b2 = (const float*)d_in[16];
    const float* wg    = (const float*)d_in[17];
    const float* bg    = (const float*)d_in[18];
    const float* wo    = (const float*)d_in[19];
    const float* bo    = (const float*)d_in[20];
    float* out = (float*)d_out;

    float* ws   = (float*)d_ws;
    float* h    = ws;                          ws += (long)NPTS * DIM;
    float* q    = ws;                          ws += (long)NPTS * 512;
    float* kb   = ws;                          ws += (long)NPTS * 128;
    float* vb   = ws;                          ws += (long)NPTS * 128;
    float* gpre = ws;                          ws += (long)NPTS * 24;
    float* ckin = ws;                          ws += (long)HKVN * NBLK * HID;
    float* cvin = ws;                          ws += (long)HKVN * NBLK * HID;
    float* hk   = ws;                          ws += (long)HKVN * NBLK * HID;
    float* hv   = ws;                          ws += (long)HKVN * NBLK * HID;
    float* ck   = ws;                          ws += (long)HKVN * NBLK * DH;
    float* cv   = ws;                          ws += (long)HKVN * NBLK * DH;
    float* csim = ws;                          ws += (long)HKVN * GN * NPTS * NBLK;
    float* comp = ws;                          ws += (long)NPTS * 512;
    float* fine = ws;                          ws += (long)NPTS * 512;
    float* loc  = ws;                          ws += (long)NPTS * 512;
    float* comb = ws;                          ws += (long)NPTS * 512;
    int*   sel  = (int*)ws;

    posemb_kernel<<<NBLK, 256, 0, stream>>>(x, pos, pe_w, pe_b, h);

    launch_gemm(h, 512, wq, 512, 0, nullptr, q, 512, NPTS, 512, 512, 0, 1.0f, stream);
    launch_gemm(h, 512, wk, 128, 0, nullptr, kb, 128, NPTS, 128, 512, 0, 1.0f, stream);
    launch_gemm(h, 512, wvw, 128, 0, nullptr, vb, 128, NPTS, 128, 512, 0, 1.0f, stream);
    launch_gemm(h, 512, wg, 24, 0, bg, gpre, 24, NPTS, 24, 512, 0, 1.0f, stream);

    build_cin<<<(HKVN * NBLK * HID + 255) / 256, 256, 0, stream>>>(kb, k_pos, ckin);
    build_cin<<<(HKVN * NBLK * HID + 255) / 256, 256, 0, stream>>>(vb, v_pos, cvin);
    launch_gemm(ckin, HID, ck_w1, HID, 0, ck_b1, hk, HID, HKVN * NBLK, HID, HID, 1, 1.0f, stream);
    launch_gemm(hk, HID, ck_w2, DH, 0, ck_b2, ck, DH, HKVN * NBLK, DH, HID, 0, 1.0f, stream);
    launch_gemm(cvin, HID, cv_w1, HID, 0, cv_b1, hv, HID, HKVN * NBLK, HID, HID, 1, 1.0f, stream);
    launch_gemm(hv, HID, cv_w2, DH, 0, cv_b2, cv, DH, HKVN * NBLK, DH, HID, 0, 1.0f, stream);

    for (int hkv = 0; hkv < HKVN; ++hkv)
        for (int g = 0; g < GN; ++g)
            launch_gemm(q + hkv * 256 + g * 64, 512,
                        ck + (long)hkv * NBLK * DH, DH, 1, nullptr,
                        csim + ((long)(hkv * GN + g) * NPTS) * NBLK, NBLK,
                        NPTS, NBLK, DH, 0, ATT_SCALE, stream);
    softmax_sel_kernel<<<HKVN * NPTS, 128, 0, stream>>>(csim, sel);

    for (int hkv = 0; hkv < HKVN; ++hkv)
        for (int g = 0; g < GN; ++g)
            launch_gemm(csim + ((long)(hkv * GN + g) * NPTS) * NBLK, NBLK,
                        cv + (long)hkv * NBLK * DH, DH, 0, nullptr,
                        comp + ((long)(hkv * GN + g) * NPTS) * DH, DH,
                        NPTS, DH, NBLK, 0, 1.0f, stream);

    fine_kernel<<<HKVN * NPTS, 128, 0, stream>>>(q, kb, vb, sel, fine);
    local_attn_wmma<<<HKVN * NBLK, 128, 0, stream>>>(q, kb, vb, loc);

    combine_kernel<<<(NPTS * 512 + 255) / 256, 256, 0, stream>>>(gpre, comp, fine, loc, comb);
    launch_gemm(comb, 512, wo, 512, 0, bo, out, 512, NPTS, 512, 512, 0, 1.0f, stream);

    (void)in_sizes; (void)n_in; (void)out_size; (void)ws_size;
}
